// AE_14542759264441
// MI455X (gfx1250) — compile-verified, matrix operand order checked by reading, not run on hardware
//
#include <hip/hip_runtime.h>

// ---------------------------------------------------------------------------
// CDNA5 / gfx1250 implementation of the recursive tree auto-encoder loss.
//
// Key observations vs. the reference:
//  * Feat[li]/Feat[ri] writes are dead; only Feat[fi] = father_f is ever
//    re-read (next level up). So decoder layer 3 only needs the 12 loss
//    columns {16..21, 32..37}, packed into ONE 16-wide WMMA tile at prep.
//  * All matmuls are 16-row tiles -> V_WMMA_F32_16X16X32_F16 (f16 in, f32 acc)
//  * One wave32 == one tile of 16 parents; layer chaining via per-wave LDS.
// ---------------------------------------------------------------------------

typedef __attribute__((ext_vector_type(16))) _Float16 v16h;
typedef __attribute__((ext_vector_type(8)))  _Float16 v8h;
typedef __attribute__((ext_vector_type(8)))  float    v8f;

#define TREE_L   21
#define NNODES   ((1 << TREE_L) - 1)
#define ROWSTR   40                      // LDS row stride in halves (80B, 16B-aligned)
#define WS_BM    64                      // byte offset of B matrices in ws
#define WS_BIAS  (64 + 7 * 1024)         // byte offset of packed biases
#define WS_FEAT  16384                   // byte offset of f16 feature buffer

__device__ __forceinline__ v8h v8h_zero() {
    v8h z;
#pragma unroll
    for (int i = 0; i < 8; ++i) z[i] = (_Float16)0.0f;
    return z;
}

__device__ __forceinline__ v8f wmma16(v16h a, v16h b, v8f c) {
    // D = A(16x32 f16) * B(32x16 f16) + C(16x16 f32)
    return __builtin_amdgcn_wmma_f32_16x16x32_f16(false, a, false, b,
                                                  (short)0, c, false, false);
}

// B matrix m, pre-swizzled into per-lane layout: Bm[m*512 + lane*16 + k]
__device__ __forceinline__ v16h loadB(const _Float16* __restrict__ Bm, int m, int lane) {
    return *(const v16h*)(Bm + m * 512 + lane * 16);
}

// Build A operand from LDS tile buf[16][ROWSTR] (row-major, K = columns 0..31).
// ISA layout: lanes 0-15 row=lane, K {0..7,16..23}; lanes 16-31 row=lane-16,
// K {8..15,24..31}.
__device__ __forceinline__ v16h load_A(const _Float16* buf, int row16, int hi) {
    const _Float16* r = buf + row16 * ROWSTR;
    v8h lo = *(const v8h*)(r + hi * 8);          // K 0..7   / 8..15
    v8h hh = *(const v8h*)(r + 16 + hi * 8);     // K 16..23 / 24..31
    v16h a;
#pragma unroll
    for (int i = 0; i < 8; ++i) { a[i] = lo[i]; a[8 + i] = hh[i]; }
    return a;
}

// Zero columns 16..31 of a tile buffer (one ds_store_b128 per lane).
__device__ __forceinline__ void zero_hi(_Float16* buf, int row16, int hi) {
    *(v8h*)(buf + row16 * ROWSTR + 16 + hi * 8) = v8h_zero();
}

// Scatter a D tile (C/D layout: lane = col, 8 rows) into LDS as f16, +bias[,ReLU]
__device__ __forceinline__ void store_D(_Float16* buf, int colBase, v8f d,
                                        const float* bias, bool relu, int lane) {
    int n  = lane & 15;
    int r0 = (lane >> 4) * 8;
    float b = bias[n];
#pragma unroll
    for (int i = 0; i < 8; ++i) {
        float v = d[i] + b;
        if (relu) v = v > 0.0f ? v : 0.0f;
        buf[(r0 + i) * ROWSTR + colBase + n] = (_Float16)v;
    }
}

// ---------------------------------------------------------------------------
// Prep: swizzle the 7 B matrices into WMMA per-lane layout + pack biases.
// mat0 enc.W1 (input permuted to [feat16 | pos5]), mat1 enc.W2, mat2 enc.W3,
// mat3 dec.W1, mat4/5 dec.W2 cols 0-15/16-31, mat6 dec.W3 packed loss cols.
// ---------------------------------------------------------------------------
__global__ void prep_weights(const float* __restrict__ ew1, const float* __restrict__ eb1,
                             const float* __restrict__ ew2, const float* __restrict__ eb2,
                             const float* __restrict__ ew3, const float* __restrict__ eb3,
                             const float* __restrict__ dw1, const float* __restrict__ db1,
                             const float* __restrict__ dw2, const float* __restrict__ db2,
                             const float* __restrict__ dw3, const float* __restrict__ db3,
                             _Float16* __restrict__ Bm, float* __restrict__ biases,
                             float* __restrict__ acc) {
    int t = blockIdx.x * blockDim.x + threadIdx.x;
    if (t == 0) acc[0] = 0.0f;
    if (t < 7 * 512) {
        int mat = t >> 9, idx = t & 511;
        int lane = idx >> 4, k = idx & 15;
        int N = lane & 15;
        int g = (lane & 16) ? 8 : 0;
        int K = (k < 8) ? (k + g) : (k + 8 + g);   // 0..31
        float v = 0.0f;
        switch (mat) {
        case 0: { // enc.W1, K layout: 0..15 = feat rows 5..20, 16..20 = pos rows 0..4
            int kk = (K < 16) ? (5 + K) : (K < 21 ? (K - 16) : -1);
            if (kk >= 0) v = ew1[kk * 16 + N];
        } break;
        case 1: if (K < 16) v = ew2[K * 16 + N]; break;
        case 2: if (K < 16) v = ew3[K * 16 + N]; break;
        case 3: if (K < 21) v = dw1[K * 16 + N]; break;  // [father_f | father_p] already
        case 4: if (K < 16) v = dw2[K * 32 + N]; break;
        case 5: if (K < 16) v = dw2[K * 32 + 16 + N]; break;
        case 6: { // dec.W3 packed: cols {16..21, 32..37}
            int sn = (N < 6) ? (16 + N) : (N < 12 ? (32 + (N - 6)) : -1);
            if (sn >= 0) v = dw3[K * 44 + sn];
        } break;
        }
        Bm[t] = (_Float16)v;
    }
    int b = t - 7 * 512;
    if (b >= 0 && b < 112) {
        float v;
        if      (b < 16)  v = eb1[b];
        else if (b < 32)  v = eb2[b - 16];
        else if (b < 48)  v = eb3[b - 32];
        else if (b < 64)  v = db1[b - 48];
        else if (b < 96)  v = db2[b - 64];
        else { int j = b - 96; v = (j < 6) ? db3[16 + j] : (j < 12 ? db3[32 + (j - 6)] : 0.0f); }
        biases[b] = v;
    }
}

// Convert leaf/all features f32 -> f16 working buffer (internal nodes are
// overwritten with father_f before they are ever read).
__global__ void feat_convert(const float* __restrict__ F, _Float16* __restrict__ out, int n) {
    int i = blockIdx.x * blockDim.x + threadIdx.x;
    if (i < n) out[i] = (_Float16)F[i];
}

// ---------------------------------------------------------------------------
// One tree level: each wave32 processes a tile of 16 parents.
// ---------------------------------------------------------------------------
__global__ void __launch_bounds__(256)
level_kernel(const float* __restrict__ X, const _Float16* __restrict__ Bm,
             const float* __restrict__ biases, _Float16* __restrict__ featWS,
             float* __restrict__ acc, int fbase, int Pn, int numTiles,
             float sXY, float sA) {
    __shared__ _Float16 lds[8 * 2 * 16 * ROWSTR];   // 8 waves x 2 tile buffers

    const int wave = threadIdx.x >> 5;
    const int lane = threadIdx.x & 31;
    const int tile = blockIdx.x * 8 + wave;
    if (tile >= numTiles) return;                   // wave-uniform exit (EXEC stays full)

    _Float16* Pb = lds + wave * (2 * 16 * ROWSTR);
    _Float16* Qb = Pb + 16 * ROWSTR;

    const int tbase = tile * 16;
    const int rv    = (Pn - tbase < 16) ? (Pn - tbase) : 16;   // valid rows
    const int row16 = lane & 15;
    const int hi    = lane >> 4;

    // --- stage children: lanes 0-15 -> left child row into Pb,
    //                     lanes 16-31 -> right child row into Qb.
    // Row layout (K): cols 0..15 = feat (f16), 16..20 = pos, 21..31 = 0.
    zero_hi(Pb, row16, hi);
    zero_hi(Qb, row16, hi);
    {
        int rc = (row16 < rv) ? row16 : (rv - 1);
        int f  = fbase + tbase + rc;
        int c  = 2 * f + 1 + hi;
        _Float16* dst = (hi ? Qb : Pb) + row16 * ROWSTR;
        const v8h* fp = (const v8h*)(featWS + (long)c * 16);
        v8h f0 = fp[0], f1 = fp[1];
        *(v8h*)(dst)     = f0;
        *(v8h*)(dst + 8) = f1;
        const float* xp = X + (long)c * 5;
#pragma unroll
        for (int j = 0; j < 5; ++j) dst[16 + j] = (_Float16)xp[j];
    }

    v8f zero = {};
    v16h aL = load_A(Pb, row16, hi);
    v16h aR = load_A(Qb, row16, hi);

    // --- encoder, left branch (21->16 relu, 16->16 relu, 16->16)
    v8f t1 = wmma16(aL, loadB(Bm, 0, lane), zero);
    zero_hi(Pb, row16, hi);                        // clear stale pos cols before K=16 reuse
    store_D(Pb, 0, t1, biases + 0, true, lane);
    v16h a = load_A(Pb, row16, hi);
    v8f t2 = wmma16(a, loadB(Bm, 1, lane), zero);
    store_D(Pb, 0, t2, biases + 16, true, lane);
    a = load_A(Pb, row16, hi);
    v8f fat = wmma16(a, loadB(Bm, 2, lane), zero);

    // --- encoder, right branch (accumulates onto 'fat' via C operand)
    t1 = wmma16(aR, loadB(Bm, 0, lane), zero);
    zero_hi(Qb, row16, hi);
    store_D(Qb, 0, t1, biases + 0, true, lane);
    a = load_A(Qb, row16, hi);
    t2 = wmma16(a, loadB(Bm, 1, lane), zero);
    store_D(Qb, 0, t2, biases + 16, true, lane);
    a = load_A(Qb, row16, hi);
    fat = wmma16(a, loadB(Bm, 2, lane), fat);

    // --- father = encL + encR (+ 2*b3, one bias per branch); stage as dec input
    zero_hi(Pb, row16, hi);
    {
        int n = lane & 15, r0 = hi * 8;
        float b = 2.0f * biases[32 + n];
#pragma unroll
        for (int i = 0; i < 8; ++i)
            Pb[(r0 + i) * ROWSTR + n] = (_Float16)(fat[i] + b);
    }
    if (!hi) {  // father pos -> cols 16..20
        int rc = (row16 < rv) ? row16 : (rv - 1);
        const float* xp = X + (long)(fbase + tbase + rc) * 5;
#pragma unroll
        for (int j = 0; j < 5; ++j) Pb[row16 * ROWSTR + 16 + j] = (_Float16)xp[j];
    }
    if (hi && row16 < rv) {  // persist father_f for the next level up
        int f = fbase + tbase + row16;
        const v8h* sp = (const v8h*)(Pb + row16 * ROWSTR);
        v8h f0 = sp[0], f1 = sp[1];
        v8h* op = (v8h*)(featWS + (long)f * 16);
        op[0] = f0; op[1] = f1;
    }

    // --- decoder: 21->16 relu, 16->32 relu, 32->12(packed loss cols)
    a = load_A(Pb, row16, hi);
    v8f u = wmma16(a, loadB(Bm, 3, lane), zero);
    zero_hi(Qb, row16, hi);
    store_D(Qb, 0, u, biases + 48, true, lane);
    a = load_A(Qb, row16, hi);
    v8f w0 = wmma16(a, loadB(Bm, 4, lane), zero);
    v8f w1 = wmma16(a, loadB(Bm, 5, lane), zero);
    store_D(Pb, 0,  w0, biases + 64, true, lane);
    store_D(Pb, 16, w1, biases + 80, true, lane);
    a = load_A(Pb, row16, hi);
    v8f o = wmma16(a, loadB(Bm, 6, lane), zero);

    // --- loss: packed cols 0,1=tanh 2,3=sigmoid 4=id (left), 6..10 same (right)
    float part = 0.0f;
    {
        int  j  = lane & 15;
        bool ls = (j < 5), rs = (j >= 6 && j < 11);
        if (ls || rs) {
            int   jq = ls ? j : (j - 6);
            float b3 = biases[96 + j];
            int   r0 = hi * 8;
#pragma unroll
            for (int i = 0; i < 8; ++i) {
                int m = r0 + i;
                if (m < rv) {
                    int f = fbase + tbase + m;
                    int c = 2 * f + 1 + (rs ? 1 : 0);
                    float q  = X[(long)c * 5 + jq];
                    float oo = o[i] + b3;
                    if (jq < 2)      { float p = tanhf(oo);               float d = p - q; part += d * d * sXY; }
                    else if (jq < 4) { float p = 1.0f / (1.0f + __expf(-oo)); float d = p - q; part += d * d * sXY; }
                    else             { part += fabsf(oo - q) * sA; }
                }
            }
        }
    }
#pragma unroll
    for (int off = 16; off > 0; off >>= 1) part += __shfl_xor(part, off, 32);
    if (lane == 0) atomicAdd(acc, part);
}

__global__ void finalize(const float* __restrict__ acc, float* __restrict__ out) {
    if (threadIdx.x == 0 && blockIdx.x == 0) {
        float L = acc[0] * (1.0f / 20.0f);   // mean over the 20 levels
        out[0] = L; out[1] = 0.0f; out[2] = L; out[3] = 0.0f;
    }
}

// ---------------------------------------------------------------------------
extern "C" void kernel_launch(void* const* d_in, const int* in_sizes, int n_in,
                              void* d_out, int out_size, void* d_ws, size_t ws_size,
                              hipStream_t stream) {
    (void)in_sizes; (void)n_in; (void)out_size; (void)ws_size;
    // Input order: X, Feature, I_list[20] (unused: indices are analytic),
    // Node_is_leaf (unused), enc(w1,b1,w2,b2,w3,b3), dec(w1,b1,w2,b2,w3,b3)
    const float* X       = (const float*)d_in[0];
    const float* Feature = (const float*)d_in[1];
    const float* ew1 = (const float*)d_in[23]; const float* eb1 = (const float*)d_in[24];
    const float* ew2 = (const float*)d_in[25]; const float* eb2 = (const float*)d_in[26];
    const float* ew3 = (const float*)d_in[27]; const float* eb3 = (const float*)d_in[28];
    const float* dw1 = (const float*)d_in[29]; const float* db1 = (const float*)d_in[30];
    const float* dw2 = (const float*)d_in[31]; const float* db2 = (const float*)d_in[32];
    const float* dw3 = (const float*)d_in[33]; const float* db3 = (const float*)d_in[34];

    char*      ws     = (char*)d_ws;
    float*     acc    = (float*)ws;
    _Float16*  Bm     = (_Float16*)(ws + WS_BM);
    float*     biases = (float*)(ws + WS_BIAS);
    _Float16*  featWS = (_Float16*)(ws + WS_FEAT);   // NNODES*16 halves (~64MB)

    prep_weights<<<15, 256, 0, stream>>>(ew1, eb1, ew2, eb2, ew3, eb3,
                                         dw1, db1, dw2, db2, dw3, db3,
                                         Bm, biases, acc);

    int total = NNODES * 16;
    feat_convert<<<(total + 255) / 256, 256, 0, stream>>>(Feature, featWS, total);

    for (int d = TREE_L - 2; d >= 0; --d) {
        int Pn    = 1 << d;
        int fbase = Pn - 1;
        int tiles = (Pn + 15) / 16;
        int blocks = (tiles + 7) / 8;
        float sXY = 1.0f / (2.0f * (float)Pn);
        float sA  = 1.0f / (float)Pn;
        level_kernel<<<blocks, 256, 0, stream>>>(X, Bm, biases, featWS, acc,
                                                 fbase, Pn, tiles, sXY, sA);
    }

    finalize<<<1, 64, 0, stream>>>(acc, (float*)d_out);
}
